// Qwen3_5GatedDeltaNet_10118942949428
// MI455X (gfx1250) — matrix operand order repaired and checked by reading, not compiled, not run on hardware
//
#include <hip/hip_runtime.h>
#include <hip/hip_bf16.h>
#include <math.h>

typedef __attribute__((ext_vector_type(16))) __bf16 v16bf;
typedef __attribute__((ext_vector_type(8)))  __bf16 v8bf;
typedef __attribute__((ext_vector_type(8)))  float  v8f;

#define T_SEQ   1024
#define HID     2048
#define HK      16
#define HV      32
#define DK      128
#define DV      128
#define KEY_DIM (HK*DK)                 // 2048
#define VAL_DIM (HV*DV)                 // 4096
#define NQKVZ   (2*KEY_DIM + 2*VAL_DIM) // 12288
#define PER_K   (2*DK + 2*2*DV)         // 768
#define CC      64                      // chunk length
#define NCHUNK  (T_SEQ/CC)

// ---------------------------------------------------------------------------
// WMMA helpers
// ---------------------------------------------------------------------------
__device__ __forceinline__ v8f wmma_bf16(v16bf a, v16bf b, v8f c) {
  return __builtin_amdgcn_wmma_f32_16x16x32_bf16(false, a, false, b,
                                                 (short)0, c, false, false);
}
// A-fragment (16x32): lanes 0-15 rows m, k in {0..15}; lanes16-31 k in {16..31}
__device__ __forceinline__ v16bf ldAfrag(const __bf16* rowp, int lane) {
  const int kh = (lane >> 4) << 4;
  v8bf lo = *(const v8bf*)(rowp + kh);
  v8bf hi = *(const v8bf*)(rowp + kh + 8);
  return __builtin_shufflevector(lo, hi, 0,1,2,3,4,5,6,7,8,9,10,11,12,13,14,15);
}
// B-fragment (32x16) from n-major [n][k] storage: lane's column, 16 contiguous k
__device__ __forceinline__ v16bf ldBfrag(const __bf16* rowp, int lane) {
  const int kh = (lane >> 4) << 4;
  v8bf lo = *(const v8bf*)(rowp + kh);
  v8bf hi = *(const v8bf*)(rowp + kh + 8);
  return __builtin_shufflevector(lo, hi, 0,1,2,3,4,5,6,7,8,9,10,11,12,13,14,15);
}

// ---------------------------------------------------------------------------
// Generic GEMM: C[M,N] = A[M,K] * B[K,N], f32 in/out, bf16 WMMA compute.
// Requires M%64==0, N%64==0, K%32==0.
// ---------------------------------------------------------------------------
#define BM 64
#define BN 64
#define BK 32
#define LDSP 40

__global__ __launch_bounds__(256)
void gemm_bf16wmma(const float* __restrict__ A, const float* __restrict__ B,
                   float* __restrict__ C, int M, int N, int Kdim) {
  __shared__ __bf16 As[2][BM][LDSP];
  __shared__ __bf16 Bs[2][BN][LDSP];   // transposed tile [n][k]

  const int tid  = threadIdx.x;
  const int lane = tid & 31;
  const int wid  = tid >> 5;
  const int wm   = wid & 3;
  const int wn   = wid >> 2;
  const int bm0  = blockIdx.y * BM;
  const int bn0  = blockIdx.x * BN;

  const int arow = tid >> 2;
  const int acol = (tid & 3) * 8;
  const int brow = tid >> 3;
  const int bcol = (tid & 7) * 8;

  const float* aptr = A + (size_t)(bm0 + arow) * Kdim + acol;
  const float* bptr = B + (size_t)brow * N + bn0 + bcol;

  auto stA = [&](int buf, float4 x0, float4 x1) {
    v8bf av = { (__bf16)x0.x,(__bf16)x0.y,(__bf16)x0.z,(__bf16)x0.w,
                (__bf16)x1.x,(__bf16)x1.y,(__bf16)x1.z,(__bf16)x1.w };
    *(v8bf*)&As[buf][arow][acol] = av;
  };
  auto stB = [&](int buf, float4 y0, float4 y1) {
    float ys[8] = { y0.x,y0.y,y0.z,y0.w, y1.x,y1.y,y1.z,y1.w };
#pragma unroll
    for (int i = 0; i < 8; ++i) Bs[buf][bcol + i][brow] = (__bf16)ys[i];
  };

  {
    float4 x0 = *(const float4*)(aptr);
    float4 x1 = *(const float4*)(aptr + 4);
    float4 y0 = *(const float4*)(bptr);
    float4 y1 = *(const float4*)(bptr + 4);
    stA(0, x0, x1); stB(0, y0, y1);
  }
  __syncthreads();

  v8f acc0 = {};
  v8f acc1 = {};
  const int nk = Kdim / BK;
  float4 na0, na1, nb0, nb1;

  for (int kt = 0; kt < nk; ++kt) {
    const int  buf  = kt & 1;
    const bool more = (kt + 1) < nk;
    if (more) {
      const float* an = aptr + (kt + 1) * BK;
      const float* bn = bptr + (size_t)(kt + 1) * BK * N;
      __builtin_prefetch(an + BK, 0, 1);
      __builtin_prefetch(bn + (size_t)BK * N, 0, 1);
      na0 = *(const float4*)(an);
      na1 = *(const float4*)(an + 4);
      nb0 = *(const float4*)(bn);
      nb1 = *(const float4*)(bn + 4);
    }
    {
      v16bf af = ldAfrag(&As[buf][wm * 16 + (lane & 15)][0], lane);
      v16bf bf0 = ldBfrag(&Bs[buf][wn * 32 + (lane & 15)][0], lane);
      acc0 = wmma_bf16(af, bf0, acc0);
      v16bf bf1 = ldBfrag(&Bs[buf][wn * 32 + 16 + (lane & 15)][0], lane);
      acc1 = wmma_bf16(af, bf1, acc1);
    }
    __syncthreads();
    if (more) { stA(buf ^ 1, na0, na1); stB(buf ^ 1, nb0, nb1); }
    __syncthreads();
  }

  const int crow0 = bm0 + wm * 16 + ((lane >> 4) << 3);
  const int ccol0 = bn0 + wn * 32 + (lane & 15);
#pragma unroll
  for (int r = 0; r < 8; ++r) {
    C[(size_t)(crow0 + r) * N + ccol0]      = acc0[r];
    C[(size_t)(crow0 + r) * N + ccol0 + 16] = acc1[r];
  }
}

// ---------------------------------------------------------------------------
// Depthwise causal conv (K=4) + SiLU + per-head L2 norm for q/k.
// ---------------------------------------------------------------------------
__global__ __launch_bounds__(128)
void conv_silu_norm(const float* __restrict__ qkvz,
                    const float* __restrict__ conv_w,
                    const float* __restrict__ conv_b,
                    const float* __restrict__ conv_state,
                    float* __restrict__ qc, float* __restrict__ kc,
                    float* __restrict__ vc) {
  const int t = blockIdx.y;
  const int g = blockIdx.x;
  const int d = threadIdx.x;

  int cc, col, kind;
  if (g < 16)      { kind = 0; int hk = g;      cc = hk * 128 + d;
                     col = hk * PER_K + d; }
  else if (g < 32) { kind = 1; int hk = g - 16; cc = KEY_DIM + hk * 128 + d;
                     col = hk * PER_K + DK + d; }
  else             { kind = 2; int hv = g - 32; cc = 2 * KEY_DIM + hv * 128 + d;
                     col = (hv >> 1) * PER_K + 2 * DK + (hv & 1) * DV + d; }

  float acc = conv_b[cc];
#pragma unroll
  for (int j = 0; j < 4; ++j) {
    const int tau = t + j;
    float x = (tau < 3) ? conv_state[cc * 3 + tau]
                        : qkvz[(size_t)(tau - 3) * NQKVZ + col];
    acc += conv_w[cc * 4 + j] * x;
  }
  const float y = acc / (1.0f + __expf(-acc));

  if (kind == 2) {
    vc[((size_t)t * HV + (g - 32)) * DV + d] = y;
    return;
  }
  __shared__ float red[4];
  float ss = y * y;
#pragma unroll
  for (int off = 16; off > 0; off >>= 1) ss += __shfl_xor(ss, off);
  if ((threadIdx.x & 31) == 0) red[threadIdx.x >> 5] = ss;
  __syncthreads();
  const float tot = red[0] + red[1] + red[2] + red[3];
  const float inv = rsqrtf(tot + 1e-6f);
  if (kind == 0)
    qc[((size_t)t * HK + g) * DK + d] = y * inv * 0.08838834764831845f;
  else
    kc[((size_t)t * HK + (g - 16)) * DK + d] = y * inv;
}

// ---------------------------------------------------------------------------
// Chunked gated delta-rule scan (UT transform), WMMA throughout.
// One block per value-head (32 blocks), 256 threads = 8 waves.
// State S (128x128 f32) lives in WMMA accumulators: wave w owns rows
// [16w,16w+16) x all 128 cols = 8 v8f accumulators.
// ---------------------------------------------------------------------------
__global__ __launch_bounds__(256)
void chunk_scan_gated(const float* __restrict__ qc, const float* __restrict__ kc,
                      const float* __restrict__ vc, const float* __restrict__ ba,
                      const float* __restrict__ A_log,
                      const float* __restrict__ dt_bias,
                      const float* __restrict__ rms_scale,
                      const float* __restrict__ qkvz,
                      const float* __restrict__ rstate,
                      float* __restrict__ gated) {
  __shared__ __bf16 S_T[128][136];   // state, B-layout [n=dv][k=dk]
  __shared__ __bf16 Kb[CC][136];     // plain K chunk [t][dk]
  __shared__ __bf16 Qb[CC][136];     // plain Q chunk [t][dk]
  __shared__ __bf16 Xb[128][72];     // B-layout [n][k=t]: rhs, then U
  __shared__ __bf16 KT[128][72];     // A-layout [dk][t]: K^T * decay
  __shared__ float  Pm[CC][CC + 1];  // strict-lower beta*decay*KK^T
  __shared__ float  Tm[CC][CC + 1];  // (I+Pm)^{-1}
  __shared__ __bf16 Ab[CC][72];      // bf16 of Tm, then masked Att
  __shared__ float  gsh[CC], cum[CC], apos[CC], wk[CC], beta_sh[CC];
  __shared__ float  rsums[2][4][2][8];

  const int h    = blockIdx.x;
  const int tid  = threadIdx.x;
  const int lane = tid & 31;
  const int wid  = tid >> 5;
  const int l16  = lane & 15;
  const int lh   = lane >> 4;
  const int wm   = wid & 3;     // 64-row GEMMs: row tile
  const int wn   = wid >> 2;    // 64-row GEMMs: col half
  const int hk   = h >> 1;
  const int zbase = hk * PER_K + 2 * DK + 2 * DV + (h & 1) * DV;
  const int bcol = hk * 4 + (h & 1);
  const int acol = hk * 4 + 2 + (h & 1);

  const float nA  = -__expf(A_log[h]);
  const float dtb = dt_bias[h];

  // persistent state accumulators: rows 16*wid .. +16, cols all 128
  v8f Sacc[8];
#pragma unroll
  for (int j = 0; j < 8; ++j) {
#pragma unroll
    for (int r = 0; r < 8; ++r) {
      const int m = wid * 16 + lh * 8 + r;
      const int n = j * 16 + l16;
      Sacc[j][r] = rstate[((size_t)h * DK + m) * DV + n];
    }
  }
  const v8f vzero = {0.f,0.f,0.f,0.f,0.f,0.f,0.f,0.f};

  for (int c = 0; c < NCHUNK; ++c) {
    const int t0 = c * CC;

    // ---- stage 1: rebuild S_T (bf16, transposed) + load K/Q chunk + scalars
#pragma unroll
    for (int j = 0; j < 8; ++j)
#pragma unroll
      for (int r = 0; r < 8; ++r)
        S_T[j * 16 + l16][wid * 16 + lh * 8 + r] = (__bf16)Sacc[j][r];

    {
      const int i  = tid >> 2;
      const int c0 = (tid & 3) * 32;
      const float* kr = &kc[((size_t)(t0 + i) * HK + hk) * DK + c0];
      const float* qr = &qc[((size_t)(t0 + i) * HK + hk) * DK + c0];
#pragma unroll
      for (int u = 0; u < 4; ++u) {
        float4 x0 = *(const float4*)(kr + u * 8);
        float4 x1 = *(const float4*)(kr + u * 8 + 4);
        v8bf kv = { (__bf16)x0.x,(__bf16)x0.y,(__bf16)x0.z,(__bf16)x0.w,
                    (__bf16)x1.x,(__bf16)x1.y,(__bf16)x1.z,(__bf16)x1.w };
        *(v8bf*)&Kb[i][c0 + u * 8] = kv;
        float4 y0 = *(const float4*)(qr + u * 8);
        float4 y1 = *(const float4*)(qr + u * 8 + 4);
        v8bf qv = { (__bf16)y0.x,(__bf16)y0.y,(__bf16)y0.z,(__bf16)y0.w,
                    (__bf16)y1.x,(__bf16)y1.y,(__bf16)y1.z,(__bf16)y1.w };
        *(v8bf*)&Qb[i][c0 + u * 8] = qv;
      }
    }
    if (tid < CC) {
      const int t = t0 + tid;
      const float bv = ba[t * 2 * HV + bcol];
      const float av = ba[t * 2 * HV + acol];
      beta_sh[tid] = 1.0f / (1.0f + __expf(-bv));
      const float xx = av + dtb;
      gsh[tid] = nA * ((xx > 20.0f) ? xx : log1pf(__expf(xx)));
    }
    __syncthreads();
    if (tid == 0) {
      float s = 0.f;
      for (int i = 0; i < CC; ++i) { s += gsh[i]; cum[i] = s; }
    }
    __syncthreads();
    if (tid < CC) {
      apos[tid] = __expf(cum[tid]);                // <= 1
      wk[tid]   = __expf(cum[CC - 1] - cum[tid]);  // <= 1
    }
    __syncthreads();

    // ---- stage 2: R = K@S0 ; rhs = beta*(V - apos*R) -> Xb ; P = K@K^T
    {
      v8f racc[4] = { vzero, vzero, vzero, vzero };
#pragma unroll
      for (int ks = 0; ks < 4; ++ks) {
        v16bf af = ldAfrag(&Kb[wm * 16 + l16][ks * 32], lane);
#pragma unroll
        for (int cn = 0; cn < 4; ++cn)
          racc[cn] = wmma_bf16(af,
              ldBfrag(&S_T[wn * 64 + cn * 16 + l16][ks * 32], lane), racc[cn]);
      }
#pragma unroll
      for (int cn = 0; cn < 4; ++cn)
#pragma unroll
        for (int r = 0; r < 8; ++r) {
          const int m = wm * 16 + lh * 8 + r;
          const int n = wn * 64 + cn * 16 + l16;
          const float vv = vc[((size_t)(t0 + m) * HV + h) * DV + n];
          Xb[n][m] = (__bf16)(beta_sh[m] * (vv - apos[m] * racc[cn][r]));
        }
      v8f pacc[2] = { vzero, vzero };
#pragma unroll
      for (int ks = 0; ks < 4; ++ks) {
        v16bf af = ldAfrag(&Kb[wm * 16 + l16][ks * 32], lane);
#pragma unroll
        for (int cn = 0; cn < 2; ++cn)
          pacc[cn] = wmma_bf16(af,
              ldBfrag(&Kb[wn * 32 + cn * 16 + l16][ks * 32], lane), pacc[cn]);
      }
#pragma unroll
      for (int cn = 0; cn < 2; ++cn)
#pragma unroll
        for (int r = 0; r < 8; ++r) {
          const int i = wm * 16 + lh * 8 + r;
          const int j = wn * 32 + cn * 16 + l16;
          Pm[i][j] = (j < i)
              ? beta_sh[i] * __expf(cum[i] - cum[j]) * pacc[cn][r] : 0.f;
        }
    }
    __syncthreads();

    // ---- stage 3: forward substitution T = (I+Pm)^{-1}
    for (int i = 0; i < CC; ++i) {
      if (tid < CC) {
        float s = (i == tid) ? 1.f : 0.f;
        for (int j = 0; j < i; ++j) s -= Pm[i][j] * Tm[j][tid];
        Tm[i][tid] = s;
      }
      __syncthreads();
    }
    {
      const int i  = tid >> 2;
      const int j0 = (tid & 3) * 16;
#pragma unroll
      for (int j = 0; j < 16; ++j) Ab[i][j0 + j] = (__bf16)Tm[i][j0 + j];
    }
    __syncthreads();

    // ---- stage 4: U = T @ rhs  (then overwrite Xb with U)
    {
      v8f uacc[4] = { vzero, vzero, vzero, vzero };
#pragma unroll
      for (int ks = 0; ks < 2; ++ks) {
        v16bf af = ldAfrag(&Ab[wm * 16 + l16][ks * 32], lane);
#pragma unroll
        for (int cn = 0; cn < 4; ++cn)
          uacc[cn] = wmma_bf16(af,
              ldBfrag(&Xb[wn * 64 + cn * 16 + l16][ks * 32], lane), uacc[cn]);
      }
      __syncthreads();
#pragma unroll
      for (int cn = 0; cn < 4; ++cn)
#pragma unroll
        for (int r = 0; r < 8; ++r)
          Xb[wn * 64 + cn * 16 + l16][wm * 16 + lh * 8 + r] =
              (__bf16)uacc[cn][r];
    }
    __syncthreads();

    // ---- stage 5: Att = mask(Q@K^T * decay) -> Ab
    {
      v8f aacc[2] = { vzero, vzero };
#pragma unroll
      for (int ks = 0; ks < 4; ++ks) {
        v16bf af = ldAfrag(&Qb[wm * 16 + l16][ks * 32], lane);
#pragma unroll
        for (int cn = 0; cn < 2; ++cn)
          aacc[cn] = wmma_bf16(af,
              ldBfrag(&Kb[wn * 32 + cn * 16 + l16][ks * 32], lane), aacc[cn]);
      }
      __syncthreads();   // Ab (=T) fully consumed by stage 4
#pragma unroll
      for (int cn = 0; cn < 2; ++cn)
#pragma unroll
        for (int r = 0; r < 8; ++r) {
          const int i = wm * 16 + lh * 8 + r;
          const int j = wn * 32 + cn * 16 + l16;
          Ab[i][j] = (__bf16)((j <= i)
              ? __expf(cum[i] - cum[j]) * aacc[cn][r] : 0.f);
        }
    }
    __syncthreads();

    // ---- stage 6: O = apos*(Q@S0) + Att@U ; fused RMS + silu(z) gating
    {
      v8f oacc[4] = { vzero, vzero, vzero, vzero };
#pragma unroll
      for (int ks = 0; ks < 4; ++ks) {
        v16bf af = ldAfrag(&Qb[wm * 16 + l16][ks * 32], lane);
#pragma unroll
        for (int cn = 0; cn < 4; ++cn)
          oacc[cn] = wmma_bf16(af,
              ldBfrag(&S_T[wn * 64 + cn * 16 + l16][ks * 32], lane), oacc[cn]);
      }
#pragma unroll
      for (int cn = 0; cn < 4; ++cn)
#pragma unroll
        for (int r = 0; r < 8; ++r)
          oacc[cn][r] = oacc[cn][r] * apos[wm * 16 + lh * 8 + r];
#pragma unroll
      for (int ks = 0; ks < 2; ++ks) {
        v16bf af = ldAfrag(&Ab[wm * 16 + l16][ks * 32], lane);
#pragma unroll
        for (int cn = 0; cn < 4; ++cn)
          oacc[cn] = wmma_bf16(af,
              ldBfrag(&Xb[wn * 64 + cn * 16 + l16][ks * 32], lane), oacc[cn]);
      }
      // per-row sum of squares: reduce over cn, then lanes 0..15 (xor 1,2,4,8)
#pragma unroll
      for (int r = 0; r < 8; ++r) {
        float p = 0.f;
#pragma unroll
        for (int cn = 0; cn < 4; ++cn) p += oacc[cn][r] * oacc[cn][r];
        p += __shfl_xor(p, 1); p += __shfl_xor(p, 2);
        p += __shfl_xor(p, 4); p += __shfl_xor(p, 8);
        if (l16 == 0) rsums[wn][wm][lh][r] = p;
      }
      __syncthreads();
#pragma unroll
      for (int cn = 0; cn < 4; ++cn)
#pragma unroll
        for (int r = 0; r < 8; ++r) {
          const int m = wm * 16 + lh * 8 + r;
          const int n = wn * 64 + cn * 16 + l16;
          const int t = t0 + m;
          const float tot = rsums[0][wm][lh][r] + rsums[1][wm][lh][r];
          const float rms = rsqrtf(tot * (1.0f / DV) + 1e-6f);
          const float zv  = qkvz[(size_t)t * NQKVZ + zbase + n];
          const float sz  = zv / (1.0f + __expf(-zv));
          gated[((size_t)t * HV + h) * DV + n] =
              oacc[cn][r] * rms * rms_scale[n] * sz;
        }
    }

    // ---- stage 7: state update  S = gamma*S + (K^T*decay) @ U
    {
      const int mm = tid >> 1;
      const int k0 = (tid & 1) * 32;
#pragma unroll
      for (int j = 0; j < 32; ++j)
        KT[mm][k0 + j] = (__bf16)((float)Kb[k0 + j][mm] * wk[k0 + j]);
    }
    __syncthreads();
    {
      const float gam = __expf(cum[CC - 1]);
#pragma unroll
      for (int j = 0; j < 8; ++j)
#pragma unroll
        for (int r = 0; r < 8; ++r) Sacc[j][r] = Sacc[j][r] * gam;
#pragma unroll
      for (int ks = 0; ks < 2; ++ks) {
        v16bf af = ldAfrag(&KT[wid * 16 + l16][ks * 32], lane);
#pragma unroll
        for (int j = 0; j < 8; ++j)
          Sacc[j] = wmma_bf16(af,
              ldBfrag(&Xb[j * 16 + l16][ks * 32], lane), Sacc[j]);
      }
    }
    __syncthreads();
  }
}

// ---------------------------------------------------------------------------
extern "C" void kernel_launch(void* const* d_in, const int* in_sizes, int n_in,
                              void* d_out, int out_size, void* d_ws,
                              size_t ws_size, hipStream_t stream) {
  const float* hs     = (const float*)d_in[0];
  const float* Wqkvz  = (const float*)d_in[1];
  const float* Wba    = (const float*)d_in[2];
  const float* convw  = (const float*)d_in[3];
  const float* convb  = (const float*)d_in[4];
  const float* Alog   = (const float*)d_in[5];
  const float* dtb    = (const float*)d_in[6];
  const float* rmss   = (const float*)d_in[7];
  const float* Wout   = (const float*)d_in[8];
  const float* cstate = (const float*)d_in[9];
  const float* rstate = (const float*)d_in[10];
  float* out = (float*)d_out;

  char* w = (char*)d_ws;
  float* qkvz  = (float*)w;  w += (size_t)T_SEQ * NQKVZ * 4;
  float* ba    = (float*)w;  w += (size_t)T_SEQ * 2 * HV * 4;
  float* qc    = (float*)w;  w += (size_t)T_SEQ * HK * DK * 4;
  float* kc    = (float*)w;  w += (size_t)T_SEQ * HK * DK * 4;
  float* vc    = (float*)w;  w += (size_t)T_SEQ * HV * DV * 4;
  float* gated = (float*)w;  w += (size_t)T_SEQ * VAL_DIM * 4;

  dim3 blk(256);
  gemm_bf16wmma<<<dim3(NQKVZ / BN, T_SEQ / BM), blk, 0, stream>>>(
      hs, Wqkvz, qkvz, T_SEQ, NQKVZ, HID);
  gemm_bf16wmma<<<dim3(1, T_SEQ / BM), blk, 0, stream>>>(
      hs, Wba, ba, T_SEQ, 2 * HV, HID);
  conv_silu_norm<<<dim3(64, T_SEQ), dim3(128), 0, stream>>>(
      qkvz, convw, convb, cstate, qc, kc, vc);
  chunk_scan_gated<<<dim3(HV), dim3(256), 0, stream>>>(
      qc, kc, vc, ba, Alog, dtb, rmss, qkvz, rstate, gated);
  gemm_bf16wmma<<<dim3(HID / BN, T_SEQ / BM), blk, 0, stream>>>(
      gated, Wout, out, T_SEQ, HID, VAL_DIM);
}
